// RecurrentRGCNREGCN_82471962017814
// MI455X (gfx1250) — matrix operand order repaired and checked by reading, not compiled, not run on hardware
//
#include <hip/hip_runtime.h>
#include <math.h>

// ---------------- problem constants (match reference) ----------------
#define N_NODES   20000
#define HDIM      512
#define R2_REL    500
#define T_STEPS   8
#define E_EDGES   100000
#define L_LEN     200000
#define SLOPE_NEG 0.2291666666666667f   // (1/8 + 1/3)/2

// ---------------- WMMA / vector types ----------------
typedef __attribute__((ext_vector_type(16))) __bf16        v16bf;
typedef __attribute__((ext_vector_type(8)))  float         v8f;
typedef __attribute__((ext_vector_type(4)))  unsigned int  u32x4;
typedef __attribute__((ext_vector_type(8)))  int           i32x8;
typedef __attribute__((ext_vector_type(4)))  int           i32x4;

union FragBF { v16bf v; uint4 u[2]; };

__device__ __forceinline__ unsigned short f2bf(float f) {
  unsigned int u = __float_as_uint(f);
  return (unsigned short)((u + 0x7FFFu + ((u >> 16) & 1u)) >> 16); // RNE
}
__device__ __forceinline__ float sigmoidf_(float x) { return 1.0f / (1.0f + __expf(-x)); }

// ==================================================================
// Tensor Data Mover staging (gfx1250). Falls back to cooperative
// synchronous copies if the builtin is absent in this toolchain.
// ==================================================================
#if __has_builtin(__builtin_amdgcn_tensor_load_to_lds)
#define USE_TDM 1
#else
#define USE_TDM 0
#endif

#if __has_builtin(__builtin_amdgcn_s_wait_tensorcnt)
#define WAIT_TENSORCNT(n) __builtin_amdgcn_s_wait_tensorcnt((short)(n))
#else
#define WAIT_TENSORCNT(n) asm volatile("s_wait_tensorcnt %0" :: "i"(n) : "memory")
#endif

__device__ __forceinline__ unsigned lds_off(const void* p) {
  return (unsigned)(unsigned long long)(size_t)p;   // low 32 bits of LDS generic addr
}

#if USE_TDM
// 2D TDM: copy [tileRows x 32] bf16 tile (row stride = strideElems elements)
// into LDS, padding each 64B row with 16B -> LDS row stride 40 ushorts.
// Rows/cols beyond (rowsRemain, colsRemain) are hardware zero-filled.
__device__ __forceinline__ void tdm_load_2d(unsigned ldsAddr, const unsigned short* gptr,
                                            int rowsRemain, int colsRemain,
                                            int strideElems, int tileRows) {
  const unsigned long long ga = (unsigned long long)(size_t)gptr;
  u32x4 g0;
  g0[0] = 1u;                                              // count=1 (user descriptor)
  g0[1] = ldsAddr;                                         // lds_addr
  g0[2] = (unsigned)(ga & 0xFFFFFFFFull);                  // global_addr[31:0]
  g0[3] = (unsigned)((ga >> 32) & 0x01FFFFFFull) | (2u << 30); // addr[56:32] | type=2
  i32x8 g1;
  g1[0] = (int)((1u << 16)        // data_size = 1 -> 2 bytes
              | (1u << 20)        // pad_enable
              | (3u << 22)        // pad_interval: every 16 DWORDs (one 64B row)
              | (3u << 25));      // pad_amount: 4 DWORDs (16B)
  g1[1] = (int)(((unsigned)colsRemain & 0xFFFFu) << 16);                 // tdim0[15:0]
  g1[2] = (int)((((unsigned)colsRemain >> 16) & 0xFFFFu)
              | (((unsigned)rowsRemain & 0xFFFFu) << 16));               // tdim0[31:16]|tdim1[15:0]
  g1[3] = (int)((((unsigned)rowsRemain >> 16) & 0xFFFFu) | (32u << 16)); // tdim1[31:16]|tile_dim0=32
  g1[4] = (int)(unsigned)tileRows;                                       // tile_dim1 (tile_dim2=0)
  g1[5] = (int)(unsigned)strideElems;                                    // tensor_dim0_stride
  g1[6] = 0;
  g1[7] = 0;
  const i32x4 z4 = {0, 0, 0, 0};
#if defined(__clang_major__) && (__clang_major__ >= 23)
  const i32x8 z8 = {0, 0, 0, 0, 0, 0, 0, 0};
  __builtin_amdgcn_tensor_load_to_lds(g0, g1, z4, z4, z8, 0);
#else
  __builtin_amdgcn_tensor_load_to_lds(g0, g1, z4, z4, 0);
#endif
}
#endif

// ==================================================================
// bf16 WMMA GEMM: C[M,N] (f32) = A[M,K] (bf16) x Bt[N,K]^T (bf16)
// A, Bt row-major bf16 in global (A pre-converted, Bt pre-transposed).
// degMode: 0 = store all rows; 1 = rows with deg[m]>0; 2 = rows deg[m]==0
// Block: 256 threads (8 wave32). Tile: 128(M) x 64(N) x 32(K),
// double-buffered LDS staged by the Tensor Data Mover.
// ==================================================================
#define BM 128
#define BN 64
#define BK 32
#define LDS_STR 40   // padded row stride (ushorts): 64B data + 16B pad

__device__ __forceinline__ void stage_tiles(unsigned short* AsS, unsigned short* BsS,
                                            const unsigned short* __restrict__ A,
                                            const unsigned short* __restrict__ Bt,
                                            int M, int N, int K,
                                            int m0, int n0, int k0, int tid, int wid) {
#if USE_TDM
  if (wid == 0) {
    tdm_load_2d(lds_off(AsS), A  + (size_t)m0 * K + k0, M - m0, K - k0, K, BM);
    tdm_load_2d(lds_off(BsS), Bt + (size_t)n0 * K + k0, N - n0, K - k0, K, BN);
  }
  (void)tid;
#else
  // cooperative staging: A tile = 512 uint4 chunks, B tile = 256 uint4 chunks
#pragma unroll
  for (int j = 0; j < 2; ++j) {
    const int c = tid * 2 + j;
    const int r = c >> 2, cc = c & 3;
    const int m = m0 + r;
    uint4 v = make_uint4(0u, 0u, 0u, 0u);
    if (m < M) v = *(const uint4*)(A + (size_t)m * K + k0 + cc * 8);
    *(uint4*)(AsS + r * LDS_STR + cc * 8) = v;
  }
  {
    const int r = tid >> 2, cc = tid & 3;
    const uint4 v = *(const uint4*)(Bt + (size_t)(n0 + r) * K + k0 + cc * 8);
    *(uint4*)(BsS + r * LDS_STR + cc * 8) = v;
  }
#endif
}

__global__ __launch_bounds__(256) void
k_gemm_bf16(const unsigned short* __restrict__ A, const unsigned short* __restrict__ Bt,
            float* __restrict__ C, int M, int N, int K,
            const float* __restrict__ deg, int degMode)
{
  __shared__ __align__(16) unsigned short As[2][BM * LDS_STR];
  __shared__ __align__(16) unsigned short Bs[2][BN * LDS_STR];

  const int tid  = threadIdx.x;
  const int lane = tid & 31;
  const int wid  = tid >> 5;
  const int wm   = wid & 3;        // 4 waves along M (32 rows each)
  const int wn   = wid >> 2;       // 2 waves along N (32 cols each)
  const int nl   = lane & 15;
  const int hi   = lane >> 4;
  const int m0   = blockIdx.y * BM;
  const int n0   = blockIdx.x * BN;
  const int nK   = K >> 5;

  v8f acc[2][2];
#pragma unroll
  for (int mi = 0; mi < 2; ++mi)
#pragma unroll
    for (int ni = 0; ni < 2; ++ni)
#pragma unroll
      for (int r = 0; r < 8; ++r) acc[mi][ni][r] = 0.0f;

  stage_tiles(As[0], Bs[0], A, Bt, M, N, K, m0, n0, 0, tid, wid);

  for (int ki = 0; ki < nK; ++ki) {
    const int cur = ki & 1;
    if (ki + 1 < nK)
      stage_tiles(As[cur ^ 1], Bs[cur ^ 1], A, Bt, M, N, K, m0, n0, (ki + 1) << 5, tid, wid);
#if USE_TDM
    if (wid == 0) {      // in-order TENSORcnt: <=2 outstanding => current stage landed
      if (ki + 1 < nK) WAIT_TENSORCNT(2);
      else             WAIT_TENSORCNT(0);
    }
#endif
    __syncthreads();

    const unsigned short* Ac = As[cur];
    const unsigned short* Bc = Bs[cur];
    FragBF a[2], b[2];
#pragma unroll
    for (int mi = 0; mi < 2; ++mi) {
      const int r = wm * 32 + mi * 16 + nl;                    // A row (M)
      a[mi].u[0] = *(const uint4*)(Ac + r * LDS_STR + hi * 8);       // K = hi*8..+7
      a[mi].u[1] = *(const uint4*)(Ac + r * LDS_STR + hi * 8 + 16);  // K = hi*8+16..+23
    }
#pragma unroll
    for (int ni = 0; ni < 2; ++ni) {
      const int c = wn * 32 + ni * 16 + nl;                    // B col (N)
      b[ni].u[0] = *(const uint4*)(Bc + c * LDS_STR + hi * 8);       // K = hi*8..+7
      b[ni].u[1] = *(const uint4*)(Bc + c * LDS_STR + hi * 8 + 16);  // K = hi*8+16..+23
    }
#pragma unroll
    for (int mi = 0; mi < 2; ++mi)
#pragma unroll
      for (int ni = 0; ni < 2; ++ni)
        acc[mi][ni] = __builtin_amdgcn_wmma_f32_16x16x32_bf16(
            false, a[mi].v, false, b[ni].v, (short)0, acc[mi][ni], false, false);
    __syncthreads();
  }

  // ---- epilogue: (optionally row-masked) store ----
#pragma unroll
  for (int mi = 0; mi < 2; ++mi)
#pragma unroll
    for (int ni = 0; ni < 2; ++ni) {
      const int n = n0 + wn * 32 + ni * 16 + nl;
#pragma unroll
      for (int r = 0; r < 8; ++r) {
        const int m = m0 + wm * 32 + mi * 16 + (hi << 3) + r;
        if (m < M) {
          bool st = true;
          if (degMode == 1)      st =  (deg[m] > 0.0f);
          else if (degMode == 2) st = !(deg[m] > 0.0f);
          if (st) C[(size_t)m * N + n] = acc[mi][ni][r];
        }
      }
    }
}

// ==================================================================
// Elementwise / reduction glue kernels
// ==================================================================
__global__ void k_zero(float* __restrict__ p, long long n) {
  long long i = (long long)blockIdx.x * blockDim.x + threadIdx.x;
  if (i < n) p[i] = 0.0f;
}

__global__ void k_copy(const float* __restrict__ in, float* __restrict__ out, long long n) {
  long long i = (long long)blockIdx.x * blockDim.x + threadIdx.x;
  if (i < n) out[i] = in[i];
}

__global__ void k_cvt_bf16(const float* __restrict__ in, unsigned short* __restrict__ out, int n) {
  int i = blockIdx.x * blockDim.x + threadIdx.x;
  if (i < n) out[i] = f2bf(in[i]);
}

// out[C,R] = bf16(in[R,C]^T)
__global__ void k_tcvt_bf16(const float* __restrict__ in, unsigned short* __restrict__ out,
                            int R, int Ccols) {
  int i = blockIdx.x * blockDim.x + threadIdx.x;
  if (i < R * Ccols) {
    int r = i / Ccols, c = i % Ccols;
    out[(size_t)c * R + r] = f2bf(in[i]);
  }
}

// per-row L2 normalize (in-place safe)
__global__ __launch_bounds__(256) void
k_l2n(const float* __restrict__ in, float* __restrict__ out, int Hd) {
  const int m = blockIdx.x;
  const float* row = in + (size_t)m * Hd;
  float s = 0.0f;
  for (int c = threadIdx.x; c < Hd; c += blockDim.x) { float v = row[c]; s += v * v; }
  __shared__ float red[256];
  red[threadIdx.x] = s;
  __syncthreads();
  for (int off = 128; off > 0; off >>= 1) {
    if (threadIdx.x < off) red[threadIdx.x] += red[threadIdx.x + off];
    __syncthreads();
  }
  const float inv = 1.0f / fmaxf(sqrtf(red[0]), 1e-12f);
  for (int c = threadIdx.x; c < Hd; c += blockDim.x)
    out[(size_t)m * Hd + c] = row[c] * inv;
}

// per-relation segment sum of h[rte] over L entries (block = one entry)
__global__ __launch_bounds__(128) void
k_seg(const int* __restrict__ rte, const int* __restrict__ roe,
      const float* __restrict__ h, float* __restrict__ ssum, float* __restrict__ cnt) {
  const int l = blockIdx.x;
  const int e = rte[l], r = roe[l];
  const int c = threadIdx.x << 2;
  const float4 v = *(const float4*)(h + (size_t)e * HDIM + c);
  float* dp = ssum + (size_t)r * HDIM + c;
  atomicAdd(dp + 0, v.x); atomicAdd(dp + 1, v.y);
  atomicAdd(dp + 2, v.z); atomicAdd(dp + 3, v.w);
  if (threadIdx.x == 0) atomicAdd(cnt + r, 1.0f);
}

// x_input[r] = [emb_rel[r] | x_mean[r]]
__global__ void k_xin(const float* __restrict__ emb_rel, const float* __restrict__ ssum,
                      const float* __restrict__ cnt, float* __restrict__ xin) {
  int i = blockIdx.x * blockDim.x + threadIdx.x;
  if (i >= R2_REL * HDIM) return;
  const int r = i / HDIM, c = i & (HDIM - 1);
  const float cv = cnt[r];
  const float xm = (cv > 0.0f) ? (ssum[i] / fmaxf(cv, 1.0f)) : 0.0f;
  xin[(size_t)r * (2 * HDIM) + c]        = emb_rel[i];
  xin[(size_t)r * (2 * HDIM) + HDIM + c] = xm;
}

// GRU gates (torch GRUCell math), in-place update of h0
__global__ void k_gru(const float* __restrict__ gi, const float* __restrict__ gh,
                      const float* __restrict__ bih, const float* __restrict__ bhh,
                      float* __restrict__ h0) {
  int i = blockIdx.x * blockDim.x + threadIdx.x;
  if (i >= R2_REL * HDIM) return;
  const int r = i / HDIM, c = i & (HDIM - 1);
  const float* gir = gi + (size_t)r * (3 * HDIM);
  const float* ghr = gh + (size_t)r * (3 * HDIM);
  const float ir = gir[c]            + bih[c];
  const float iz = gir[HDIM + c]     + bih[HDIM + c];
  const float in_= gir[2 * HDIM + c] + bih[2 * HDIM + c];
  const float hr = ghr[c]            + bhh[c];
  const float hz = ghr[HDIM + c]     + bhh[HDIM + c];
  const float hn = ghr[2 * HDIM + c] + bhh[2 * HDIM + c];
  const float rg = sigmoidf_(ir + hr);
  const float z  = sigmoidf_(iz + hz);
  const float n  = tanhf(in_ + rg * hn);
  h0[i] = (1.0f - z) * n + z * h0[i];
}

__global__ void k_deg(const int* __restrict__ dst, float* __restrict__ deg, int E) {
  int i = blockIdx.x * blockDim.x + threadIdx.x;
  if (i < E) atomicAdd(deg + dst[i], 1.0f);
}

// scatter-add (h[src] + emb_rel[et]) into aggsum[dst]  (block = one edge)
__global__ __launch_bounds__(128) void
k_edge_scatter(const int* __restrict__ src, const int* __restrict__ dst,
               const int* __restrict__ et, const float* __restrict__ cur,
               const float* __restrict__ emb_rel, float* __restrict__ aggsum) {
  const int e = blockIdx.x;
  const int s = src[e], d = dst[e], r = et[e];
  const int c = threadIdx.x << 2;
  const float4 hv = *(const float4*)(cur + (size_t)s * HDIM + c);
  const float4 rv = *(const float4*)(emb_rel + (size_t)r * HDIM + c);
  float* dp = aggsum + (size_t)d * HDIM + c;
  atomicAdd(dp + 0, hv.x + rv.x);
  atomicAdd(dp + 1, hv.y + rv.y);
  atomicAdd(dp + 2, hv.z + rv.z);
  atomicAdd(dp + 3, hv.w + rv.w);
}

// out = agg_lin * 1/max(deg,1) + loop ; rrelu(eval)
__global__ void k_combine(const float* __restrict__ aggl, const float* __restrict__ loop,
                          const float* __restrict__ deg, float* __restrict__ out) {
  int i = blockIdx.x * blockDim.x + threadIdx.x;
  if (i >= N_NODES * HDIM) return;
  const int m = i / HDIM;
  const float o = aggl[i] * (1.0f / fmaxf(deg[m], 1.0f)) + loop[i];
  out[i] = (o >= 0.0f) ? o : SLOPE_NEG * o;
}

// tw = sigmoid(h@Wt + b); h = tw*cur + (1-tw)*h ; also writes hist[t]
__global__ void k_timegate(const float* __restrict__ tg, const float* __restrict__ bias,
                           const float* __restrict__ cur, float* __restrict__ h,
                           float* __restrict__ hist) {
  int i = blockIdx.x * blockDim.x + threadIdx.x;
  if (i >= N_NODES * HDIM) return;
  const float tw = sigmoidf_(tg[i] + bias[i & (HDIM - 1)]);
  const float v = tw * cur[i] + (1.0f - tw) * h[i];
  h[i] = v;
  hist[i] = v;
}

// ==================================================================
// Host orchestration
// ==================================================================
extern "C" void kernel_launch(void* const* d_in, const int* in_sizes, int n_in,
                              void* d_out, int out_size, void* d_ws, size_t ws_size,
                              hipStream_t stream) {
  (void)in_sizes; (void)n_in; (void)out_size; (void)ws_size;

  const int*   src      = (const int*)d_in[0];
  const int*   dst      = (const int*)d_in[1];
  const int*   etype    = (const int*)d_in[2];
  const int*   r_to_e   = (const int*)d_in[3];
  const int*   rel_of_e = (const int*)d_in[4];
  const float* dyn_emb  = (const float*)d_in[5];
  const float* emb_rel  = (const float*)d_in[6];
  const float* w_nei    = (const float*)d_in[7];   // [2,H,H]
  const float* w_loop   = (const float*)d_in[8];   // [2,H,H]
  const float* w_evo    = (const float*)d_in[9];   // [2,H,H]
  const float* w_ih     = (const float*)d_in[10];  // [3H,2H]
  const float* w_hh     = (const float*)d_in[11];  // [3H,H]
  const float* b_ih     = (const float*)d_in[12];
  const float* b_hh     = (const float*)d_in[13];
  const float* tgw      = (const float*)d_in[14];  // [H,H]
  const float* tgb      = (const float*)d_in[15];

  float* out = (float*)d_out;

  const long long NH = (long long)N_NODES * HDIM;
  const long long RH = (long long)R2_REL * HDIM;
  const int HH = HDIM * HDIM;
  const int H3 = 3 * HDIM;

  // ---- workspace carve-out ----
  char*  ws  = (char*)d_ws;
  size_t off = 0;
  auto carve = [&](size_t bytes) -> char* {
    char* p = ws + off;
    off += (bytes + 255) & ~(size_t)255;
    return p;
  };
  float* h    = (float*)carve(NH * 4);
  float* S0   = (float*)carve(NH * 4);   // aggsum
  float* S1   = (float*)carve(NH * 4);   // agg_lin / time-gate gemm out
  float* S2   = (float*)carve(NH * 4);   // loop (wl/we merged via masked store)
  float* S3   = (float*)carve(NH * 4);   // cur (layer output)
  float* h0   = (float*)carve(RH * 4);
  float* cnt  = (float*)carve(R2_REL * 4);
  float* ssum = (float*)carve(RH * 4);
  float* xin  = (float*)carve((size_t)R2_REL * 2 * HDIM * 4);
  float* gi   = (float*)carve((size_t)R2_REL * H3 * 4);
  float* gh   = (float*)carve((size_t)R2_REL * H3 * 4);
  float* deg  = (float*)carve((size_t)N_NODES * 4);
  unsigned short* Abf    = (unsigned short*)carve((size_t)NH * 2);      // bf16 A staging
  unsigned short* wnT    = (unsigned short*)carve((size_t)2 * HH * 2);  // W^T bf16
  unsigned short* wlT    = (unsigned short*)carve((size_t)2 * HH * 2);
  unsigned short* weT    = (unsigned short*)carve((size_t)2 * HH * 2);
  unsigned short* tgwT   = (unsigned short*)carve((size_t)HH * 2);
  unsigned short* wih_bf = (unsigned short*)carve((size_t)H3 * 2 * HDIM * 2); // [3H,2H]
  unsigned short* whh_bf = (unsigned short*)carve((size_t)H3 * HDIM * 2);     // [3H,H]

  auto blks = [](long long n) { return (unsigned)((n + 255) / 256); };
  auto gemm = [&](const unsigned short* A, const unsigned short* Bt, float* C,
                  int M, int Nn, int K, const float* degp, int mode) {
    dim3 g((unsigned)(Nn / BN), (unsigned)((M + BM - 1) / BM));
    k_gemm_bf16<<<g, 256, 0, stream>>>(A, Bt, C, M, Nn, K, degp, mode);
  };

  // ---- weight prep: square weights transposed to [N][K] bf16; GRU weights are
  //      used as x@W^T so Bt = (W^T)^T = W (plain convert) ----
  for (int l = 0; l < 2; ++l) {
    k_tcvt_bf16<<<blks(HH), 256, 0, stream>>>(w_nei  + (size_t)l * HH, wnT + (size_t)l * HH, HDIM, HDIM);
    k_tcvt_bf16<<<blks(HH), 256, 0, stream>>>(w_loop + (size_t)l * HH, wlT + (size_t)l * HH, HDIM, HDIM);
    k_tcvt_bf16<<<blks(HH), 256, 0, stream>>>(w_evo  + (size_t)l * HH, weT + (size_t)l * HH, HDIM, HDIM);
  }
  k_tcvt_bf16<<<blks(HH), 256, 0, stream>>>(tgw, tgwT, HDIM, HDIM);
  k_cvt_bf16<<<blks((long long)H3 * 2 * HDIM), 256, 0, stream>>>(w_ih, wih_bf, H3 * 2 * HDIM);
  k_cvt_bf16<<<blks((long long)H3 * HDIM),     256, 0, stream>>>(w_hh, whh_bf, H3 * HDIM);

  // ---- initial state: h = l2n(dynamic_emb), h0 = emb_rel ----
  k_l2n<<<N_NODES, 256, 0, stream>>>(dyn_emb, h, HDIM);
  k_copy<<<blks(RH), 256, 0, stream>>>(emb_rel, h0, RH);

  for (int t = 0; t < T_STEPS; ++t) {
    const int* src_t = src      + (size_t)t * E_EDGES;
    const int* dst_t = dst      + (size_t)t * E_EDGES;
    const int* et_t  = etype    + (size_t)t * E_EDGES;
    const int* rte_t = r_to_e   + (size_t)t * L_LEN;
    const int* roe_t = rel_of_e + (size_t)t * L_LEN;
    float* hist_t = out + (size_t)t * NH;

    // --- relation-GRU branch ---
    k_zero<<<blks(RH), 256, 0, stream>>>(ssum, RH);
    k_zero<<<blks(R2_REL), 256, 0, stream>>>(cnt, R2_REL);
    k_seg<<<L_LEN, 128, 0, stream>>>(rte_t, roe_t, h, ssum, cnt);
    k_xin<<<blks(RH), 256, 0, stream>>>(emb_rel, ssum, cnt, xin);
    k_cvt_bf16<<<blks((long long)R2_REL * 2 * HDIM), 256, 0, stream>>>(xin, Abf, R2_REL * 2 * HDIM);
    gemm(Abf, wih_bf, gi, R2_REL, H3, 2 * HDIM, nullptr, 0);   // gi = x_input @ W_ih^T
    k_cvt_bf16<<<blks(RH), 256, 0, stream>>>(h0, Abf, (int)RH);
    gemm(Abf, whh_bf, gh, R2_REL, H3, HDIM, nullptr, 0);       // gh = h0 @ W_hh^T
    k_gru<<<blks(RH), 256, 0, stream>>>(gi, gh, b_ih, b_hh, h0);
    k_l2n<<<R2_REL, 256, 0, stream>>>(h0, h0, HDIM);

    // --- per-timestep in-degree ---
    k_zero<<<blks(N_NODES), 256, 0, stream>>>(deg, N_NODES);
    k_deg<<<blks(E_EDGES), 256, 0, stream>>>(dst_t, deg, E_EDGES);

    // --- two RGCN layers (agg-then-GEMM factorization) ---
    const float* curIn = h;
    for (int l = 0; l < 2; ++l) {
      k_zero<<<blks(NH), 256, 0, stream>>>(S0, NH);
      k_edge_scatter<<<E_EDGES, 128, 0, stream>>>(src_t, dst_t, et_t, curIn, emb_rel, S0);
      k_cvt_bf16<<<blks(NH), 256, 0, stream>>>(S0, Abf, (int)NH);
      gemm(Abf, wnT + (size_t)l * HH, S1, N_NODES, HDIM, HDIM, nullptr, 0);  // agg_lin
      k_cvt_bf16<<<blks(NH), 256, 0, stream>>>(curIn, Abf, (int)NH);
      gemm(Abf, wlT + (size_t)l * HH, S2, N_NODES, HDIM, HDIM, deg, 1);      // rows deg>0
      gemm(Abf, weT + (size_t)l * HH, S2, N_NODES, HDIM, HDIM, deg, 2);      // rows deg==0
      k_combine<<<blks(NH), 256, 0, stream>>>(S1, S2, deg, S3);
      curIn = S3;
    }
    k_l2n<<<N_NODES, 256, 0, stream>>>(S3, S3, HDIM);

    // --- time gate + state update (writes hist[t] and h) ---
    k_cvt_bf16<<<blks(NH), 256, 0, stream>>>(h, Abf, (int)NH);
    gemm(Abf, tgwT, S1, N_NODES, HDIM, HDIM, nullptr, 0);
    k_timegate<<<blks(NH), 256, 0, stream>>>(S1, tgb, S3, h, hist_t);
  }

  // evolved relation embeddings appended after hist
  k_copy<<<blks(RH), 256, 0, stream>>>(h0, out + (size_t)T_STEPS * NH, RH);
}
